// GatedLinearAttention_40458591928587
// MI455X (gfx1250) — compile-verified, matrix operand order checked by reading, not run on hardware
//
#include <hip/hip_runtime.h>
#include <cstdint>
#include <cstddef>

// ---------------------------------------------------------------------------
// Gated Linear Attention forward for MI455X (gfx1250, wave32, WMMA + TDM)
// B=2, L=2048, D=2048, H=16, dk=64, dv=128
// ---------------------------------------------------------------------------

typedef __attribute__((ext_vector_type(16))) __bf16 v16bf;
typedef __attribute__((ext_vector_type(8)))  float  v8f;
typedef __attribute__((ext_vector_type(4)))  unsigned int v4u;
typedef __attribute__((ext_vector_type(8)))  int    v8i_t;
typedef __attribute__((ext_vector_type(4)))  int    v4i_t;

union BF16Frag { v16bf v; uint32_t u[8]; };

__device__ __forceinline__ uint32_t pack_bf16(float a, float b) {
    uint32_t ua = __float_as_uint(a);
    uint32_t ub = __float_as_uint(b);
    ua = (ua + 0x7FFFu + ((ua >> 16) & 1u)) >> 16;   // round-to-nearest-even
    ub = (ub + 0x7FFFu + ((ub >> 16) & 1u)) >> 16;
    return ua | (ub << 16);
}

__device__ __forceinline__ uint16_t f32_to_bf16(float a) {
    uint32_t u = __float_as_uint(a);
    u = (u + 0x7FFFu + ((u >> 16) & 1u)) >> 16;
    return (uint16_t)u;
}

// ---------------------------------------------------------------------------
// f32 -> bf16 conversion (2 elements / thread)
// ---------------------------------------------------------------------------
__global__ void __launch_bounds__(256)
cvt_bf16_kernel(const float* __restrict__ src, uint32_t* __restrict__ dst, int n2)
{
    const int i = blockIdx.x * 256 + threadIdx.x;
    if (i < n2) {
        float2 f = ((const float2*)src)[i];
        dst[i] = pack_bf16(f.x, f.y);
    }
}

// ---------------------------------------------------------------------------
// TDM: DMA one [rows x 32] bf16 tile (row stride K elements) into LDS with
// 96-byte LDS row stride. D# packing per CDNA5 ISA 8.3-8.6.
// ---------------------------------------------------------------------------
#define TK 32
#define LDSW 24   // uint32 per LDS row (48 bf16 = 96B, 16B aligned)

__device__ __forceinline__ void tdm_load_tile(uint32_t lds_off,
                                              const uint16_t* gptr,
                                              int rows, int K)
{
    const uint64_t ga = (uint64_t)(uintptr_t)gptr;
    v4u   g0;
    v8i_t g1;
    v4i_t gz4 = {0, 0, 0, 0};
    v8i_t gz8 = {0, 0, 0, 0, 0, 0, 0, 0};

    g0[0] = 1u;                                            // count=1, user mode
    g0[1] = lds_off;                                       // lds_addr (bytes)
    g0[2] = (uint32_t)ga;                                  // global_addr[31:0]
    g0[3] = (uint32_t)((ga >> 32) & 0x1FFFFFFu) | (2u << 30);  // [56:32] | type=2

    g1[0] = (int)((1u << 16)       // data_size = 2 bytes
                | (1u << 20)       // pad_enable
                | (3u << 22)       // pad_interval: 16 DWORDs (64B)
                | (7u << 25));     // pad_amount: 8 DWORDs (32B) -> 96B stride
    g1[1] = (int)((uint32_t)(K & 0xFFFF) << 16);           // tensor_dim0 lo
    g1[2] = (int)(((uint32_t)K >> 16) | ((uint32_t)(rows & 0xFFFF) << 16));
    g1[3] = (int)(((uint32_t)rows >> 16) | ((uint32_t)TK << 16));  // tile_dim0=32
    g1[4] = rows;                                          // tile_dim1
    g1[5] = K;                                             // tensor_dim0_stride lo
    g1[6] = 0;
    g1[7] = 0;

    __builtin_amdgcn_tensor_load_to_lds(g0, g1, gz4, gz4, gz8, 0);
}

// ---------------------------------------------------------------------------
// GEMM: C[M,N] = (A[M,K] @ W[N,K]^T) * scale + bias[N]
// bf16 inputs, f32 output. 128x128 block tile, K-step 32, 256 threads
// (8 waves x (2 M-subtiles x 4 N-subtiles)). Double-buffered TDM staging.
// ---------------------------------------------------------------------------
#define BM 128
#define BN 128

__global__ void __launch_bounds__(256)
gemm_tdm(const uint16_t* __restrict__ A, const uint16_t* __restrict__ W,
         const float* __restrict__ bias, float* __restrict__ C,
         int M, int N, int K, float scale)
{
    __shared__ __align__(16) uint32_t lsA[2][BM * LDSW];
    __shared__ __align__(16) uint32_t lsW[2][BN * LDSW];

    const int tid  = threadIdx.x;
    const int lane = tid & 31;
    const int wave = tid >> 5;
    const int bm   = blockIdx.y * BM;
    const int bn   = blockIdx.x * BN;

    const int wm0 = (wave & 3) * 2;    // first of 2 M subtiles
    const int wn0 = (wave >> 2) * 4;   // first of 4 N subtiles

    const int m_half = lane >> 4;
    const int m_loc  = lane & 15;

    v8f acc[2][4];
    #pragma unroll
    for (int i = 0; i < 2; ++i)
        #pragma unroll
        for (int j = 0; j < 4; ++j)
            acc[i][j] = (v8f){0.f,0.f,0.f,0.f,0.f,0.f,0.f,0.f};

    const uint32_t offA[2] = { (uint32_t)(uintptr_t)&lsA[0][0],
                               (uint32_t)(uintptr_t)&lsA[1][0] };
    const uint32_t offW[2] = { (uint32_t)(uintptr_t)&lsW[0][0],
                               (uint32_t)(uintptr_t)&lsW[1][0] };
    const uint16_t* Abase = A + (size_t)bm * K;
    const uint16_t* Wbase = W + (size_t)bn * K;

    const int NT = K / TK;

    if (wave == 0) {                       // prologue: stage tile 0
        tdm_load_tile(offA[0], Abase, BM, K);
        tdm_load_tile(offW[0], Wbase, BN, K);
    }

    for (int kt = 0; kt < NT; ++kt) {
        const int cur = kt & 1;
        if (wave == 0) {
            if (kt + 1 < NT) {             // stage next tile, wait for current
                tdm_load_tile(offA[cur ^ 1], Abase + (kt + 1) * TK, BM, K);
                tdm_load_tile(offW[cur ^ 1], Wbase + (kt + 1) * TK, BN, K);
                __builtin_amdgcn_s_wait_tensorcnt(2);
            } else {
                __builtin_amdgcn_s_wait_tensorcnt(0);
            }
        }
        __syncthreads();

        // fragments per CDNA5 wave32 16-bit A/B layout
        BF16Frag af[2], bf[4];
        #pragma unroll
        for (int i = 0; i < 2; ++i) {
            const uint32_t* ra = &lsA[cur][((wm0 + i) * 16 + m_loc) * LDSW + 4 * m_half];
            #pragma unroll
            for (int t = 0; t < 4; ++t) { af[i].u[t] = ra[t]; af[i].u[t + 4] = ra[8 + t]; }
        }
        #pragma unroll
        for (int j = 0; j < 4; ++j) {
            const uint32_t* rb = &lsW[cur][((wn0 + j) * 16 + m_loc) * LDSW + 4 * m_half];
            #pragma unroll
            for (int t = 0; t < 4; ++t) { bf[j].u[t] = rb[t]; bf[j].u[t + 4] = rb[8 + t]; }
        }

        #pragma unroll
        for (int i = 0; i < 2; ++i)
            #pragma unroll
            for (int j = 0; j < 4; ++j)
                acc[i][j] = __builtin_amdgcn_wmma_f32_16x16x32_bf16(
                                false, af[i].v, false, bf[j].v,
                                (short)0, acc[i][j], false, false);
        __syncthreads();
    }

    // epilogue: D layout lane = N within subtile, VGPR r -> M = r + 8*half
    #pragma unroll
    for (int j = 0; j < 4; ++j) {
        const int col = bn + (wn0 + j) * 16 + m_loc;
        const float bv = bias ? bias[col] : 0.f;
        #pragma unroll
        for (int i = 0; i < 2; ++i) {
            const int row0 = bm + (wm0 + i) * 16 + 8 * m_half;
            #pragma unroll
            for (int r = 0; r < 8; ++r)
                C[(size_t)(row0 + r) * N + col] = acc[i][j][r] * scale + bv;
        }
    }
}

// ---------------------------------------------------------------------------
// Low-rank gate stage 1: T[M,16] = x[M,K] @ Wkg1[16,K]^T
// ---------------------------------------------------------------------------
__global__ void __launch_bounds__(256)
lowrank1_kernel(const float* __restrict__ x, const float* __restrict__ Wkg1,
                float* __restrict__ T, int K)
{
    const int tid = threadIdx.x;
    const int row = blockIdx.x * 16 + (tid >> 4);
    const int j   = tid & 15;
    const float* xr = x + (size_t)row * K;
    const float* wr = Wkg1 + (size_t)j * K;
    float acc = 0.f;
    for (int c = 0; c < K; c += 4) {
        float4 xv = *(const float4*)(xr + c);
        float4 wv = *(const float4*)(wr + c);
        acc += xv.x * wv.x + xv.y * wv.y + xv.z * wv.z + xv.w * wv.w;
    }
    T[(size_t)row * 16 + j] = acc;
}

// ---------------------------------------------------------------------------
// Low-rank gate stage 2: eg[M,1024] = exp(log_sigmoid(T@Wkg2^T + bkg2)/16)
// ---------------------------------------------------------------------------
__global__ void __launch_bounds__(256)
gate2_kernel(const float* __restrict__ T, const float* __restrict__ Wkg2,
             const float* __restrict__ bkg2, float* __restrict__ eg)
{
    const int idx = blockIdx.x * 256 + threadIdx.x;
    const int row = idx >> 10;
    const int n   = idx & 1023;
    const float* tr = T + (size_t)row * 16;
    const float* wr = Wkg2 + (size_t)n * 16;
    float acc = bkg2[n];
    #pragma unroll
    for (int j = 0; j < 16; ++j) acc += tr[j] * wr[j];
    float ls = fminf(acc, 0.f) - log1pf(expf(-fabsf(acc)));   // stable log_sigmoid
    eg[idx] = expf(ls * (1.f / 16.f));
}

// ---------------------------------------------------------------------------
// Chunked GLA recurrence (chunk C=64), one block per (b,h), 256 threads.
// All chunk matmuls run on WMMA:
//   cum_t = prod_{s<=t} eg_s ; Qs = q*cum ; Ks = k/cum ; K2_s = k_s*(ctot/cum_s)
//   A = tril(Qs @ Ks^T)                       [64x64,  Kc=dk=64]
//   O = Qs @ S_in^T  +  A @ V                 [64x128, Kc=64]
//   S_out^T = ctot (x) S_in^T + Vt @ K2t^T    [128x64, Kc=t=64]
// Operands stored in LDS as [rows][contraction] bf16, 144B row pitch, so both
// A- and B-fragments use the same ds_load_b128 gather as the GEMM.
// ---------------------------------------------------------------------------
#define CP32 36    // u32 pitch (144B) for 64-wide bf16 rows
#define CP16 72    // u16 pitch

__device__ __forceinline__ void frag_ld(const uint32_t* ls, int row, int kstep,
                                        int m_half, BF16Frag& f) {
    const uint32_t* p = ls + row * CP32 + kstep * 16 + 4 * m_half;
    #pragma unroll
    for (int t = 0; t < 4; ++t) { f.u[t] = p[t]; f.u[t + 4] = p[8 + t]; }
}

__global__ void __launch_bounds__(256)
gla_chunked(const float* __restrict__ q, const float* __restrict__ k,
            const float* __restrict__ eg, const float* __restrict__ v,
            float* __restrict__ o, int L)
{
    const int b    = blockIdx.x >> 4;
    const int h    = blockIdx.x & 15;
    const int tid  = threadIdx.x;
    const int lane = tid & 31;
    const int wave = tid >> 5;
    const int m_half = lane >> 4;
    const int m_loc  = lane & 15;

    __shared__ __align__(16) uint32_t Qs [64  * CP32];   //  9 KB [t][k]
    __shared__ __align__(16) uint32_t Ks [64  * CP32];   //  9 KB [s][k]
    __shared__ __align__(16) uint32_t Am [64  * CP32];   //  9 KB [t][s]
    __shared__ __align__(16) uint32_t K2t[64  * CP32];   //  9 KB [k][t]
    __shared__ __align__(16) uint32_t Vt [128 * CP32];   // 18 KB [v][t]
    __shared__ __align__(16) uint32_t Stb[128 * CP32];   // 18 KB bf16 S^T [v][k]
    __shared__ float St[128 * 64];                       // 32 KB f32  S^T [v][k]
    __shared__ float stot[64];                           // ctot per k-channel

    // zero initial state
    for (int i = tid; i < 128 * 64; i += 256) St[i] = 0.f;
    for (int i = tid; i < 128 * CP32; i += 256) Stb[i] = 0u;
    __syncthreads();

    const size_t qkBase = ((size_t)b * L) * 1024 + (size_t)h * 64;
    const size_t vBase  = ((size_t)b * L) * 2048 + (size_t)h * 128;
    const v8f vzero = (v8f){0.f,0.f,0.f,0.f,0.f,0.f,0.f,0.f};

    const int NC = L / 64;
    for (int c = 0; c < NC; ++c) {
        const int t0 = c * 64;

        // ---- stage: gate walks (threads 0..63) + V transpose (64..191) ----
        if (tid < 64) {
            const int kd = tid;
            const float* qp = q  + qkBase + (size_t)t0 * 1024 + kd;
            const float* kp = k  + qkBase + (size_t)t0 * 1024 + kd;
            const float* ep = eg + qkBase + (size_t)t0 * 1024 + kd;
            float cprod = 1.f;                      // prefix product of eg
            for (int t = 0; t < 64; ++t) {
                float e  = ep[(size_t)t * 1024];
                float qv = qp[(size_t)t * 1024];
                float kv = kp[(size_t)t * 1024];
                cprod *= e;
                ((uint16_t*)Qs)[t * CP16 + kd] = f32_to_bf16(qv * cprod);
                ((uint16_t*)Ks)[t * CP16 + kd] = f32_to_bf16(kv / cprod);
            }
            float suf = 1.f;                        // suffix product of eg
            for (int t = 63; t >= 0; --t) {
                float kv = kp[(size_t)t * 1024];
                ((uint16_t*)K2t)[kd * CP16 + t] = f32_to_bf16(kv * suf);
                suf *= ep[(size_t)t * 1024];
            }
            stot[kd] = suf;                         // = ctot
        } else if (tid < 192) {
            const int vd = tid - 64;
            const float* vp = v + vBase + (size_t)t0 * 2048 + vd;
            for (int tp = 0; tp < 32; ++tp) {
                float v0 = vp[(size_t)(2 * tp)     * 2048];
                float v1 = vp[(size_t)(2 * tp + 1) * 2048];
                Vt[vd * CP32 + tp] = pack_bf16(v0, v1);
            }
        }
        __syncthreads();

        // ---- A = tril(Qs @ Ks^T), masked, stored bf16 as [t][s] ----
        {
            const int tm  = wave & 3;
            const int sn0 = (wave >> 2) * 2;
            #pragma unroll
            for (int j = 0; j < 2; ++j) {
                v8f accA = vzero;
                #pragma unroll
                for (int ks = 0; ks < 2; ++ks) {
                    BF16Frag a, bb;
                    frag_ld(Qs, tm * 16 + m_loc, ks, m_half, a);
                    frag_ld(Ks, (sn0 + j) * 16 + m_loc, ks, m_half, bb);
                    accA = __builtin_amdgcn_wmma_f32_16x16x32_bf16(
                               false, a.v, false, bb.v, (short)0, accA, false, false);
                }
                const int s = (sn0 + j) * 16 + m_loc;
                #pragma unroll
                for (int r = 0; r < 8; ++r) {
                    const int t = tm * 16 + r + 8 * m_half;
                    const float val = (s <= t) ? accA[r] : 0.f;   // causal mask
                    ((uint16_t*)Am)[t * CP16 + s] = f32_to_bf16(val);
                }
            }
        }
        __syncthreads();

        // ---- O = Qs @ Stb^T + Am @ Vt^T, written f32 to global ----
        {
            const int tm  = wave & 3;
            const int vn0 = (wave >> 2) * 4;
            #pragma unroll
            for (int j = 0; j < 4; ++j) {
                v8f accO = vzero;
                #pragma unroll
                for (int ks = 0; ks < 2; ++ks) {      // inter-chunk: Kc = dk
                    BF16Frag a, bb;
                    frag_ld(Qs,  tm * 16 + m_loc, ks, m_half, a);
                    frag_ld(Stb, (vn0 + j) * 16 + m_loc, ks, m_half, bb);
                    accO = __builtin_amdgcn_wmma_f32_16x16x32_bf16(
                               false, a.v, false, bb.v, (short)0, accO, false, false);
                }
                #pragma unroll
                for (int ks = 0; ks < 2; ++ks) {      // intra-chunk: Kc = s
                    BF16Frag a, bb;
                    frag_ld(Am, tm * 16 + m_loc, ks, m_half, a);
                    frag_ld(Vt, (vn0 + j) * 16 + m_loc, ks, m_half, bb);
                    accO = __builtin_amdgcn_wmma_f32_16x16x32_bf16(
                               false, a.v, false, bb.v, (short)0, accO, false, false);
                }
                const int vcol = (vn0 + j) * 16 + m_loc;
                #pragma unroll
                for (int r = 0; r < 8; ++r) {
                    const int t = tm * 16 + r + 8 * m_half;
                    o[vBase + (size_t)(t0 + t) * 2048 + vcol] = accO[r];
                }
            }
        }
        __syncthreads();

        // ---- state: S^T = ctot(x)S^T + Vt @ K2t^T ----
        {
            const int vm0 = (wave & 3) * 2;
            const int kn0 = (wave >> 2) * 2;
            #pragma unroll
            for (int i = 0; i < 2; ++i) {
                #pragma unroll
                for (int j = 0; j < 2; ++j) {
                    v8f accS = vzero;
                    #pragma unroll
                    for (int ks = 0; ks < 2; ++ks) {  // Kc = t
                        BF16Frag a, bb;
                        frag_ld(Vt,  (vm0 + i) * 16 + m_loc, ks, m_half, a);
                        frag_ld(K2t, (kn0 + j) * 16 + m_loc, ks, m_half, bb);
                        accS = __builtin_amdgcn_wmma_f32_16x16x32_bf16(
                                   false, a.v, false, bb.v, (short)0, accS, false, false);
                    }
                    const int kcol = (kn0 + j) * 16 + m_loc;
                    const float tot = stot[kcol];
                    #pragma unroll
                    for (int r = 0; r < 8; ++r) {
                        const int vrow = (vm0 + i) * 16 + r + 8 * m_half;
                        const float sv = tot * St[vrow * 64 + kcol] + accS[r];
                        St[vrow * 64 + kcol] = sv;
                        ((uint16_t*)Stb)[vrow * CP16 + kcol] = f32_to_bf16(sv);
                    }
                }
            }
        }
        __syncthreads();
    }
}

// ---------------------------------------------------------------------------
// Per-head LayerNorm (no affine) fused with SiLU gating; writes bf16 so the
// output projection consumes bf16 directly.
// ---------------------------------------------------------------------------
__global__ void __launch_bounds__(128)
ln_gate_kernel(const float* __restrict__ o, const float* __restrict__ gp,
               uint16_t* __restrict__ gated)
{
    const int grp = blockIdx.x;          // row*16 + h
    const int row = grp >> 4;
    const int h   = grp & 15;
    const int j   = threadIdx.x;

    __shared__ float red[128];
    __shared__ float red2[128];

    const size_t idx = (size_t)row * 2048 + (size_t)h * 128 + j;
    const float xv = o[idx];
    red[j]  = xv;
    red2[j] = xv * xv;
    __syncthreads();
    #pragma unroll
    for (int s = 64; s > 0; s >>= 1) {
        if (j < s) { red[j] += red[j + s]; red2[j] += red2[j + s]; }
        __syncthreads();
    }
    const float mu  = red[0] * (1.f / 128.f);
    const float var = red2[0] * (1.f / 128.f) - mu * mu;
    const float nrm = (xv - mu) * rsqrtf(var + 1e-5f);

    const float g  = gp[idx];
    const float sg = g / (1.f + expf(-g));   // silu(g)
    gated[idx] = f32_to_bf16(nrm * sg);
}

// ---------------------------------------------------------------------------
// Host-side launcher
// ---------------------------------------------------------------------------
extern "C" void kernel_launch(void* const* d_in, const int* in_sizes, int n_in,
                              void* d_out, int out_size, void* d_ws, size_t ws_size,
                              hipStream_t stream) {
    (void)n_in; (void)out_size; (void)ws_size;

    const float* x    = (const float*)d_in[0];
    const float* Wq   = (const float*)d_in[1];
    const float* Wk   = (const float*)d_in[2];
    const float* Wkg1 = (const float*)d_in[3];
    const float* Wkg2 = (const float*)d_in[4];
    const float* bkg2 = (const float*)d_in[5];
    const float* Wv   = (const float*)d_in[6];
    const float* Wg   = (const float*)d_in[7];
    const float* bg   = (const float*)d_in[8];
    const float* Wo   = (const float*)d_in[9];
    float* out = (float*)d_out;

    const int D = 2048, L = 2048, H = 16;
    const int M = in_sizes[0] / D;               // B*L = 4096
    const int B = M / L;
    const float SCALE = 0.08838834764831845f;    // 128^-0.5

    const size_t MB = 1u << 20;
    uint8_t* ws = (uint8_t*)d_ws;
    float*    q   = (float*)(ws);                 // 16 MB [M,1024] f32
    float*    kk  = (float*)(ws + 16  * MB);      // 16 MB
    float*    eg  = (float*)(ws + 32  * MB);      // 16 MB
    float*    v   = (float*)(ws + 48  * MB);      // 32 MB [M,2048]
    float*    gp  = (float*)(ws + 80  * MB);      // 32 MB
    float*    o   = (float*)(ws + 112 * MB);      // 32 MB
    float*    tlow= (float*)(ws + 144 * MB);      // 256 KB [M,16]
    uint16_t* xb  = (uint16_t*)(ws + 145 * MB);   // 16 MB bf16 x
    uint16_t* wqb = (uint16_t*)(ws + 161 * MB);   //  4 MB
    uint16_t* wkb = (uint16_t*)(ws + 165 * MB);   //  4 MB
    uint16_t* wvb = (uint16_t*)(ws + 169 * MB);   //  8 MB
    uint16_t* wgb = (uint16_t*)(ws + 177 * MB);   //  8 MB
    uint16_t* wob = (uint16_t*)(ws + 185 * MB);   //  8 MB
    uint16_t* gatedb = (uint16_t*)ws;             // reuse q/kk (dead after recurrence)

    dim3 blk(256);
    auto cvt = [&](const float* s, uint16_t* d, size_t n) {
        int n2 = (int)(n / 2);
        cvt_bf16_kernel<<<(n2 + 255) / 256, blk, 0, stream>>>(s, (uint32_t*)d, n2);
    };

    // one-shot bf16 conversion (halves all GEMM read traffic)
    cvt(x,  xb,  (size_t)M * D);
    cvt(Wq, wqb, (size_t)1024 * D);
    cvt(Wk, wkb, (size_t)1024 * D);
    cvt(Wv, wvb, (size_t)2048 * D);
    cvt(Wg, wgb, (size_t)2048 * D);
    cvt(Wo, wob, (size_t)2048 * D);

    // projections: TDM-staged bf16 WMMA GEMMs
    gemm_tdm<<<dim3(1024 / BN, M / BM), blk, 0, stream>>>(xb, wqb, nullptr, q,  M, 1024, D, 1.f);
    gemm_tdm<<<dim3(1024 / BN, M / BM), blk, 0, stream>>>(xb, wkb, nullptr, kk, M, 1024, D, SCALE);
    gemm_tdm<<<dim3(2048 / BN, M / BM), blk, 0, stream>>>(xb, wvb, nullptr, v,  M, 2048, D, 1.f);
    gemm_tdm<<<dim3(2048 / BN, M / BM), blk, 0, stream>>>(xb, wgb, bg,      gp, M, 2048, D, 1.f);

    // low-rank gate path
    lowrank1_kernel<<<M / 16, blk, 0, stream>>>(x, Wkg1, tlow, D);
    gate2_kernel<<<(M * 1024) / 256, blk, 0, stream>>>(tlow, Wkg2, bkg2, eg);

    // chunked WMMA recurrence (32 workgroups, one per (b,h))
    gla_chunked<<<B * H, blk, 0, stream>>>(q, kk, eg, v, o, L);

    // per-head LayerNorm + SiLU gating -> bf16
    ln_gate_kernel<<<M * H, dim3(128), 0, stream>>>(o, gp, gatedb);

    // output projection
    gemm_tdm<<<dim3(2048 / BN, M / BM), blk, 0, stream>>>(gatedb, wob, nullptr, out, M, 2048, D, 1.f);
}
